// MSU_66503273611576
// MI455X (gfx1250) — matrix operand order repaired and checked by reading, not compiled, not run on hardware
//
#include <hip/hip_runtime.h>

// ---------------------------------------------------------------------------
// Problem constants (match the reference)
// ---------------------------------------------------------------------------
#define BB    4
#define SS    2048
#define DD    1024
#define HH    8
#define LRc   512
#define LRHc  64
#define SHDc  128
#define EXPc  2
#define KK    4
#define CONVc 1042           // 2*LR + 2*H + EXP
#define NPROJ 1554           // CONV + LR
#define BS    (BB*SS)        // 8192

// ---------------------------------------------------------------------------
// Types for WMMA
// ---------------------------------------------------------------------------
typedef __attribute__((ext_vector_type(16))) __bf16 v16bf;
typedef __attribute__((ext_vector_type(8)))  float  v8f;

union Frag16 { uint4 q[2]; v16bf v; };

static __device__ __forceinline__ unsigned short f32_to_bf16(float f) {
    unsigned int u = __float_as_uint(f);
    unsigned int r = (u + 0x7FFFu + ((u >> 16) & 1u)) >> 16;   // RNE
    return (unsigned short)r;
}

// A fragment (16x32 bf16): lane holds row m = lane&15,
// K-chunks {kb..kb+7} and {kb+16..kb+23}, kb = 8*(lane>=16).
static __device__ __forceinline__ v16bf frag_a(const unsigned short* A, int ldk,
                                               int mbase, int kk, int lane) {
    int m  = mbase + (lane & 15);
    int kb = kk * 32 + ((lane >> 4) << 3);
    const unsigned short* p = A + (size_t)m * ldk + kb;
    Frag16 f;
    f.q[0] = *reinterpret_cast<const uint4*>(p);
    f.q[1] = *reinterpret_cast<const uint4*>(p + 16);
    return f.v;
}

// B fragment (32x16 bf16) from an (N,K)-row-major matrix:
// lane holds col n, K-chunks {kb..kb+7} and {kb+16..kb+23}, kb = kk*32 + 8*(lane>=16).
static __device__ __forceinline__ v16bf frag_b(const unsigned short* Wt, int ldk,
                                               int n, int kk, int lane) {
    int kb = kk * 32 + ((lane >> 4) << 3);
    const unsigned short* p = Wt + (size_t)n * ldk + kb;
    Frag16 f;
    f.q[0] = *reinterpret_cast<const uint4*>(p);
    f.q[1] = *reinterpret_cast<const uint4*>(p + 16);
    return f.v;
}

static __device__ __forceinline__ v8f wmma_bf16(v16bf a, v16bf b, v8f c) {
    return __builtin_amdgcn_wmma_f32_16x16x32_bf16(false, a, false, b,
                                                   (short)0, c, false, false);
}

// ---------------------------------------------------------------------------
// Async global -> LDS copy (gfx1250 GLOBAL_LOAD_ASYNC_TO_LDS_B128, ASYNCcnt).
// Signature (from hipcc diagnostic): param0 = AS1 pointer to
// 'int __attribute__((vector_size(16)))' (non-const), then LDS ptr, offset, cpol.
// ---------------------------------------------------------------------------
#if defined(__has_builtin)
#if __has_builtin(__builtin_amdgcn_global_load_async_to_lds_b128)
#define HAVE_ASYNC_LDS 1
#endif
#endif

typedef int v4i_a __attribute__((vector_size(16)));

static __device__ __forceinline__ void async_cp16(unsigned short* l,
                                                  const unsigned short* g) {
#if defined(HAVE_ASYNC_LDS)
    __builtin_amdgcn_global_load_async_to_lds_b128(
        (__attribute__((address_space(1))) v4i_a*)(uintptr_t)g,
        (__attribute__((address_space(3))) v4i_a*)(uintptr_t)(unsigned)(uintptr_t)l,
        0, 0);
#else
    *reinterpret_cast<uint4*>(l) = *reinterpret_cast<const uint4*>(g);
#endif
}

static __device__ __forceinline__ void wait_async0() {
#if defined(HAVE_ASYNC_LDS)
#if __has_builtin(__builtin_amdgcn_s_wait_asynccnt)
    __builtin_amdgcn_s_wait_asynccnt(0);
#else
    asm volatile("s_wait_asynccnt 0x0" ::: "memory");
#endif
#endif
}

// ---------------------------------------------------------------------------
// fp32 -> bf16 conversion
// ---------------------------------------------------------------------------
__global__ void cvt_bf16_kernel(const float* __restrict__ src,
                                unsigned short* __restrict__ dst, int n) {
    int i = blockIdx.x * 256 + threadIdx.x;
    if (i < n) dst[i] = f32_to_bf16(src[i]);
}

// ---------------------------------------------------------------------------
// LDS-staged bf16 WMMA GEMM: C(MxN,f32) = A(MxK,bf16) * Wt(NxK,bf16)^T
// Block tile 128(M) x 64(N); 8 waves, each 16x64 (4 accumulators).
// K consumed in 32-deep steps; A(128x32)+B(64x32) double-buffered in LDS,
// filled with async copies (B shared by all waves).
// M % 128 == 0, K % 32 == 0; N may be ragged (loads clamped, stores guarded).
// ---------------------------------------------------------------------------
__global__ void gemm_bf16_kernel(const unsigned short* __restrict__ A,
                                 const unsigned short* __restrict__ Wt,
                                 float* __restrict__ C,
                                 int M, int N, int K, int ntiles) {
    __shared__ __align__(16) unsigned short Ab[2][128 * 32];
    __shared__ __align__(16) unsigned short Bb[2][64 * 32];

    int tid  = threadIdx.x;
    int lane = tid & 31;
    int wave = tid >> 5;
    int bid  = blockIdx.x;
    int mb   = (bid / ntiles) * 128;
    int nb   = (bid - (bid / ntiles) * ntiles) * 64;

    // per-thread staging assignments (3 x 16B chunks per thread per K-step)
    int qa0 = tid * 2, qa1 = tid * 2 + 1;            // A: 512 chunks
    int ra0 = qa0 >> 2, sa0 = qa0 & 3;
    int ra1 = qa1 >> 2, sa1 = qa1 & 3;
    int rb  = tid >> 2, sb = tid & 3;                // B: 256 chunks
    int nB  = nb + rb; if (nB >= N) nB = N - 1;      // clamp ragged N

    const unsigned short* gA0 = A  + (size_t)(mb + ra0) * K + sa0 * 8;
    const unsigned short* gA1 = A  + (size_t)(mb + ra1) * K + sa1 * 8;
    const unsigned short* gB  = Wt + (size_t)nB * K + sb * 8;

    int ksteps = K >> 5;

    // prologue: stage kk = 0 into buffer 0
    async_cp16(&Ab[0][qa0 * 8], gA0);
    async_cp16(&Ab[0][qa1 * 8], gA1);
    async_cp16(&Bb[0][tid * 8], gB);
    wait_async0();
    __syncthreads();

    v8f z = {0.f,0.f,0.f,0.f,0.f,0.f,0.f,0.f};
    v8f acc[4] = {z, z, z, z};

    int cur = 0;
    for (int kk = 0; kk < ksteps; ++kk) {
        int nxt = cur ^ 1;
        if (kk + 1 < ksteps) {
            int ko = (kk + 1) * 32;
            async_cp16(&Ab[nxt][qa0 * 8], gA0 + ko);
            async_cp16(&Ab[nxt][qa1 * 8], gA1 + ko);
            async_cp16(&Bb[nxt][tid * 8], gB + ko);
        }
        v16bf a = frag_a(&Ab[cur][0], 32, wave * 16, 0, lane);
#pragma unroll
        for (int j = 0; j < 4; ++j) {
            v16bf b = frag_b(&Bb[cur][0], 32, j * 16 + (lane & 15), 0, lane);
            acc[j] = wmma_bf16(a, b, acc[j]);
        }
        wait_async0();
        __syncthreads();
        cur = nxt;
    }

#pragma unroll
    for (int j = 0; j < 4; ++j) {
        int n = nb + j * 16 + (lane & 15);
        if (n < N) {
#pragma unroll
            for (int r = 0; r < 8; ++r) {
                int m = mb + wave * 16 + r + ((lane >> 4) << 3);
                C[(size_t)m * N + n] = acc[j][r];
            }
        }
    }
}

// ---------------------------------------------------------------------------
// Depthwise causal conv (K=4) + SiLU + RMS norms + expansion -> xv, fv, rin, Cg
// One workgroup per (b,s). proj is L2-resident (51MB < 192MB L2).
// ---------------------------------------------------------------------------
__global__ void conv_prep_kernel(const float* __restrict__ proj,
                                 const float* __restrict__ conv_w,
                                 const float* __restrict__ w_in_norm,
                                 const float* __restrict__ w_f_norm,
                                 const float* __restrict__ w_r_norm,
                                 float* __restrict__ xv, float* __restrict__ fv,
                                 float* __restrict__ rin_o, float* __restrict__ cg_o) {
    __shared__ float y[CONVc];
    __shared__ float red[256];
    int bs = blockIdx.x;
    int b  = bs / SS, s = bs - b * SS;
    int tid = threadIdx.x;

    for (int c = tid; c < CONVc; c += 256) {
        float acc = 0.f;
#pragma unroll
        for (int k = 0; k < KK; ++k) {
            int sr = s - (KK - 1) + k;
            if (sr >= 0)
                acc += proj[((size_t)b * SS + sr) * NPROJ + c] * conv_w[c * KK + k];
        }
        y[c] = acc / (1.f + __expf(-acc));   // silu
    }
    __syncthreads();

    float s0 = 0.f, s1 = 0.f;
    for (int c = tid; c < LRc; c += 256) { float v = y[c];       s0 += v * v; }
    for (int c = tid; c < LRc; c += 256) { float v = y[LRc + c]; s1 += v * v; }

    red[tid] = s0; __syncthreads();
    for (int off = 128; off > 0; off >>= 1) { if (tid < off) red[tid] += red[tid + off]; __syncthreads(); }
    float inv_i = rsqrtf(red[0] / (float)LRc + 1e-6f);
    __syncthreads();

    red[tid] = s1; __syncthreads();
    for (int off = 128; off > 0; off >>= 1) { if (tid < off) red[tid] += red[tid + off]; __syncthreads(); }
    float inv_f = rsqrtf(red[0] / (float)LRc + 1e-6f);
    __syncthreads();

    float sr2 = (tid < HH) ? y[2 * LRc + tid] * y[2 * LRc + tid] : 0.f;
    red[tid] = sr2; __syncthreads();
    for (int off = 128; off > 0; off >>= 1) { if (tid < off) red[tid] += red[tid + off]; __syncthreads(); }
    float inv_r = rsqrtf(red[0] / (float)HH + 1e-6f);

    float e0 = y[2 * LRc + 2 * HH];
    float e1 = y[2 * LRc + 2 * HH + 1];

    size_t base = (size_t)bs * (HH * SHDc);
    for (int i = tid; i < HH * SHDc; i += 256) {
        int hh  = i >> 7;
        int rem = i & 127;
        int ex  = rem >> 6;
        int j   = rem & 63;
        float ev   = ex ? e1 : e0;
        int   ci   = hh * LRHc + j;
        float inpn = y[ci]       * inv_i * w_in_norm[ci];
        float finn = y[LRc + ci] * inv_f * w_f_norm[ci];
        xv[base + i] = inpn * ev;
        fv[base + i] = finn * ev;
    }
    if (tid < HH) {
        rin_o[(size_t)bs * HH + tid] = y[2 * LRc + tid] * inv_r * w_r_norm[tid];
        cg_o[(size_t)bs * HH + tid]  = y[2 * LRc + HH + tid];
    }
}

// ---------------------------------------------------------------------------
// Normalize state_w matrices (Frobenius norm), transpose to (e,d), cast bf16.
// ---------------------------------------------------------------------------
__global__ void norm_state_kernel(const float* __restrict__ sw,
                                  unsigned short* __restrict__ swt) {
    __shared__ float red[256];
    int m   = blockIdx.x;
    int tid = threadIdx.x;
    const float* src = sw + (size_t)m * (SHDc * SHDc);
    float ss = 0.f;
    for (int i = tid; i < SHDc * SHDc; i += 256) { float v = src[i]; ss += v * v; }
    red[tid] = ss; __syncthreads();
    for (int off = 128; off > 0; off >>= 1) { if (tid < off) red[tid] += red[tid + off]; __syncthreads(); }
    float scale = 1.f / fmaxf(sqrtf(red[0]), 1e-12f);
    for (int i = tid; i < SHDc * SHDc; i += 256) {
        int e = i >> 7, d = i & 127;                       // out[e][d] = in[d][e]
        swt[(size_t)m * (SHDc * SHDc) + i] = f32_to_bf16(src[d * SHDc + e] * scale);
    }
}

// ---------------------------------------------------------------------------
// Sequential recurrence. One workgroup per head (8 WGs, 256 threads, 8 waves).
// Rw/Fw/W (bf16, transposed) resident in LDS. State h, gates f, and step
// inputs xv/fv live in registers (each element is private to one thread);
// next step's inputs are prefetched one full step ahead to hide latency.
// Per step: phase1 h@[Rw|Fw] (16 N-tiles, 2/wave), phase2 (r*h)@W (8 tiles).
// Fuses out = res_w*xv + h*Cg, stored bf16 for the following WMMA GEMM.
// ---------------------------------------------------------------------------
__global__ void recurrence_kernel(const unsigned short* __restrict__ swt,
                                  const float* __restrict__ xv,
                                  const float* __restrict__ fv,
                                  const float* __restrict__ rin,
                                  const float* __restrict__ cg,
                                  const float* __restrict__ res_w,
                                  unsigned short* __restrict__ out16) {
    extern __shared__ char smem[];
    unsigned short* Rwt = (unsigned short*)(smem + 0);        // 128x128 bf16
    unsigned short* Fwt = (unsigned short*)(smem + 32768);
    unsigned short* Wt  = (unsigned short*)(smem + 65536);
    unsigned short* hb  = (unsigned short*)(smem + 98304);    // 16x128 bf16
    unsigned short* rh  = (unsigned short*)(smem + 102400);   // 16x128 bf16
    float* pre1 = (float*)(smem + 106496);                    // 4x256
    float* pre2 = (float*)(smem + 110592);                    // 4x128
    float* rinL = (float*)(smem + 112640);                    // 4
    float* cgL  = (float*)(smem + 112656);                    // 4

    int tid  = threadIdx.x;
    int lane = tid & 31;
    int wave = tid >> 5;
    int hid  = blockIdx.x;
    float rw_h = res_w[hid];

    const unsigned short* srcW = swt + (size_t)hid          * (SHDc * SHDc);
    const unsigned short* srcF = swt + (size_t)(HH + hid)   * (SHDc * SHDc);
    const unsigned short* srcR = swt + (size_t)(2*HH + hid) * (SHDc * SHDc);
    for (int i = tid; i < SHDc * SHDc; i += 256) {
        Wt[i] = srcW[i]; Fwt[i] = srcF[i]; Rwt[i] = srcR[i];
    }
    for (int i = tid; i < 16 * SHDc; i += 256) { hb[i] = 0; rh[i] = 0; }

    // per-thread element ownership: i0 = tid, i1 = tid+256  -> (b, e)
    int i0 = tid, i1 = tid + 256;
    int b0 = i0 >> 7, e0 = i0 & 127;
    int b1 = i1 >> 7, e1 = i1 & 127;
    size_t a0 = ((size_t)b0 * SS * HH + hid) * SHDc + e0;   // + t*1024
    size_t a1 = ((size_t)b1 * SS * HH + hid) * SHDc + e1;
    size_t ar = (size_t)(tid < 4 ? tid : 0) * SS * HH + hid; // + t*8

    // prologue: load step 0 inputs into registers
    float xc0 = xv[a0], xc1 = xv[a1];
    float fc0 = fv[a0], fc1 = fv[a1];
    float rc = 0.f, cc = 0.f;
    if (tid < 4) { rc = rin[ar]; cc = cg[ar]; }
    float hc0 = 0.f, hc1 = 0.f;      // fp32 state, register-resident
    __syncthreads();

    v8f z = {0.f,0.f,0.f,0.f,0.f,0.f,0.f,0.f};

    for (int t = 0; t < SS; ++t) {
        // publish this step's rin/cg; prefetch next step's scalars
        if (tid < 4) {
            rinL[tid] = rc; cgL[tid] = cc;
            if (t + 1 < SS) {
                rc = rin[ar + (size_t)(t + 1) * HH];
                cc = cg [ar + (size_t)(t + 1) * HH];
            }
        }
        // prefetch next step's xv/fv (consumed one iteration later)
        float xn0 = 0.f, xn1 = 0.f, fn0 = 0.f, fn1 = 0.f;
        if (t + 1 < SS) {
            size_t o = (size_t)(t + 1) * (HH * SHDc);
            xn0 = xv[a0 + o]; xn1 = xv[a1 + o];
            fn0 = fv[a0 + o]; fn1 = fv[a1 + o];
        }

        // phase 1: pre1 = h @ [Rw | Fw]   (N = 256, 16 tiles, 2 per wave)
        {
            v8f p0 = z, p1 = z;
            int n0 = 2 * wave, n1 = 2 * wave + 1;
            const unsigned short* m0 = (n0 < 8) ? Rwt : Fwt; int t0 = n0 & 7;
            const unsigned short* m1 = (n1 < 8) ? Rwt : Fwt; int t1 = n1 & 7;
#pragma unroll
            for (int kk = 0; kk < 4; ++kk) {
                v16bf a  = frag_a(hb, SHDc, 0, kk, lane);
                v16bf bq0 = frag_b(m0, SHDc, t0 * 16 + (lane & 15), kk, lane);
                p0 = wmma_bf16(a, bq0, p0);
                v16bf bq1 = frag_b(m1, SHDc, t1 * 16 + (lane & 15), kk, lane);
                p1 = wmma_bf16(a, bq1, p1);
            }
#pragma unroll
            for (int r = 0; r < 8; ++r) {
                int m = r + ((lane >> 4) << 3);
                if (m < 4) {
                    pre1[m * 256 + n0 * 16 + (lane & 15)] = p0[r];
                    pre1[m * 256 + n1 * 16 + (lane & 15)] = p1[r];
                }
            }
        }
        __syncthreads();

        // phase 2a: gates r (consumed now), f (kept in regs); rh = bf16(r*h)
        float fg0, fg1;
        {
            float rv0 = 1.f / (1.f + __expf(-(rinL[b0] + pre1[b0 * 256 + e0])));
            fg0 = 1.f / (1.f + __expf(-(fc0 + pre1[b0 * 256 + 128 + e0])));
            rh[i0] = f32_to_bf16(rv0 * hc0);
            float rv1 = 1.f / (1.f + __expf(-(rinL[b1] + pre1[b1 * 256 + e1])));
            fg1 = 1.f / (1.f + __expf(-(fc1 + pre1[b1 * 256 + 128 + e1])));
            rh[i1] = f32_to_bf16(rv1 * hc1);
        }
        __syncthreads();

        // phase 2b: pre2 = (r*h) @ W   (N = 128, 1 tile per wave)
        {
            v8f ac = z;
#pragma unroll
            for (int kk = 0; kk < 4; ++kk) {
                v16bf a = frag_a(rh, SHDc, 0, kk, lane);
                v16bf b = frag_b(Wt, SHDc, wave * 16 + (lane & 15), kk, lane);
                ac = wmma_bf16(a, b, ac);
            }
#pragma unroll
            for (int r = 0; r < 8; ++r) {
                int m = r + ((lane >> 4) << 3);
                if (m < 4) pre2[m * 128 + wave * 16 + (lane & 15)] = ac[r];
            }
        }
        __syncthreads();

        // phase 3: state update + fused output, rotate prefetch registers
        {
            float ht0 = tanhf(xc0 + pre2[i0]);
            hc0 = fg0 * hc0 + (1.f - fg0) * ht0;
            hb[i0] = f32_to_bf16(hc0);
            out16[a0 + (size_t)t * (HH * SHDc)] = f32_to_bf16(rw_h * xc0 + hc0 * cgL[b0]);

            float ht1 = tanhf(xc1 + pre2[i1]);
            hc1 = fg1 * hc1 + (1.f - fg1) * ht1;
            hb[i1] = f32_to_bf16(hc1);
            out16[a1 + (size_t)t * (HH * SHDc)] = f32_to_bf16(rw_h * xc1 + hc1 * cgL[b1]);
        }
        xc0 = xn0; xc1 = xn1; fc0 = fn0; fc1 = fn1;
        __syncthreads();
    }
}

// ---------------------------------------------------------------------------
// Gating + RMS: u2 = bf16(rms(u * silu(gate)) * w_g_norm). One WG per (b,s).
// ---------------------------------------------------------------------------
__global__ void gate_rms_kernel(const float* __restrict__ U,
                                const float* __restrict__ proj,
                                const float* __restrict__ w_g_norm,
                                unsigned short* __restrict__ U2) {
    __shared__ float red[256];
    int bs = blockIdx.x, tid = threadIdx.x;
    const float* urow = U + (size_t)bs * LRc;
    const float* grow = proj + (size_t)bs * NPROJ + CONVc;
    float vals[2];
    float ss = 0.f;
#pragma unroll
    for (int k = 0; k < 2; ++k) {
        int i = tid + k * 256;
        float g = grow[i];
        float u = urow[i] * (g / (1.f + __expf(-g)));
        vals[k] = u;
        ss += u * u;
    }
    red[tid] = ss; __syncthreads();
    for (int off = 128; off > 0; off >>= 1) { if (tid < off) red[tid] += red[tid + off]; __syncthreads(); }
    float inv = rsqrtf(red[0] / (float)LRc + 1e-6f);
#pragma unroll
    for (int k = 0; k < 2; ++k) {
        int i = tid + k * 256;
        U2[(size_t)bs * LRc + i] = f32_to_bf16(vals[k] * inv * w_g_norm[i]);
    }
}

// ---------------------------------------------------------------------------
// Host launcher
// ---------------------------------------------------------------------------
extern "C" void kernel_launch(void* const* d_in, const int* in_sizes, int n_in,
                              void* d_out, int out_size, void* d_ws, size_t ws_size,
                              hipStream_t stream) {
    const float* x         = (const float*)d_in[0];
    const float* w_in      = (const float*)d_in[1];
    const float* conv_w    = (const float*)d_in[2];
    const float* state_w   = (const float*)d_in[3];
    const float* up_w      = (const float*)d_in[4];
    const float* out_w     = (const float*)d_in[5];
    const float* res_w     = (const float*)d_in[6];
    const float* w_in_norm = (const float*)d_in[7];
    const float* w_f_norm  = (const float*)d_in[8];
    const float* w_r_norm  = (const float*)d_in[9];
    const float* w_g_norm  = (const float*)d_in[10];
    float* outp = (float*)d_out;

    char* ws = (char*)d_ws;
    constexpr size_t SZ_X16    = (size_t)BS * DD * 2;
    constexpr size_t SZ_WIN16  = (size_t)NPROJ * DD * 2;
    constexpr size_t SZ_PROJ   = (size_t)BS * NPROJ * 4;
    constexpr size_t SZ_XV     = (size_t)BS * HH * SHDc * 4;
    constexpr size_t SZ_RIN    = (size_t)BS * HH * 4;
    constexpr size_t SZ_SWT    = (size_t)3 * HH * SHDc * SHDc * 2;
    constexpr size_t SZ_OUT16  = (size_t)BS * HH * SHDc * 2;
    constexpr size_t SZ_UPW16  = (size_t)LRHc * SHDc * 2;
    constexpr size_t SZ_OUTW16 = (size_t)DD * LRc * 2;
    constexpr size_t SZ_U      = (size_t)BS * LRc * 4;

    size_t off = 0;
    auto alloc = [&](size_t sz) { size_t o = off; off += (sz + 255) & ~(size_t)255; return o; };
    unsigned short* X16    = (unsigned short*)(ws + alloc(SZ_X16));
    unsigned short* WIN16  = (unsigned short*)(ws + alloc(SZ_WIN16));
    float*          PROJ   = (float*)         (ws + alloc(SZ_PROJ));
    float*          XV     = (float*)         (ws + alloc(SZ_XV));
    float*          FV     = (float*)         (ws + alloc(SZ_XV));
    float*          RIN    = (float*)         (ws + alloc(SZ_RIN));
    float*          CG     = (float*)         (ws + alloc(SZ_RIN));
    unsigned short* SWT    = (unsigned short*)(ws + alloc(SZ_SWT));
    unsigned short* OUT16  = (unsigned short*)(ws + alloc(SZ_OUT16));
    unsigned short* UPW16  = (unsigned short*)(ws + alloc(SZ_UPW16));
    unsigned short* OUTW16 = (unsigned short*)(ws + alloc(SZ_OUTW16));
    float*          U      = (float*)         (ws + alloc(SZ_U));
    unsigned short* U2     = (unsigned short*)(ws + alloc((size_t)BS * LRc * 2));
    (void)ws_size; (void)n_in; (void)in_sizes; (void)out_size;

    // 1) fp32 -> bf16 conversions
    {
        int n;
        n = BS * DD;       cvt_bf16_kernel<<<(n + 255) / 256, 256, 0, stream>>>(x, X16, n);
        n = NPROJ * DD;    cvt_bf16_kernel<<<(n + 255) / 256, 256, 0, stream>>>(w_in, WIN16, n);
        n = LRHc * SHDc;   cvt_bf16_kernel<<<(n + 255) / 256, 256, 0, stream>>>(up_w, UPW16, n);
        n = DD * LRc;      cvt_bf16_kernel<<<(n + 255) / 256, 256, 0, stream>>>(out_w, OUTW16, n);
    }

    // 2) proj = x @ w_in^T  (M=8192, N=1554, K=1024)
    {
        int ntiles = (NPROJ + 63) / 64;                    // 25
        gemm_bf16_kernel<<<(BS / 128) * ntiles, 256, 0, stream>>>(
            X16, WIN16, PROJ, BS, NPROJ, DD, ntiles);
    }

    // 3) conv + silu + rms + expand
    conv_prep_kernel<<<BS, 256, 0, stream>>>(PROJ, conv_w, w_in_norm, w_f_norm,
                                             w_r_norm, XV, FV, RIN, CG);

    // 4) normalize + transpose state matrices
    norm_state_kernel<<<3 * HH, 256, 0, stream>>>(state_w, SWT);

    // 5) sequential recurrence (8 workgroups, one per head), ~110KB dynamic LDS
    {
        constexpr int SMEM_D = 112672;
        (void)hipFuncSetAttribute((const void*)recurrence_kernel,
                                  hipFuncAttributeMaxDynamicSharedMemorySize, SMEM_D);
        recurrence_kernel<<<HH, 256, SMEM_D, stream>>>(SWT, XV, FV, RIN, CG,
                                                       res_w, OUT16);
    }

    // 6) U = out @ up_w^T  (M = B*S*H = 65536, N=64, K=128)
    {
        int M = BS * HH;
        gemm_bf16_kernel<<<(M / 128) * 1, 256, 0, stream>>>(
            OUT16, UPW16, U, M, LRHc, SHDc, 1);
    }

    // 7) gating + rms -> U2 (bf16)
    gate_rms_kernel<<<BS, 256, 0, stream>>>(U, PROJ, w_g_norm, U2);

    // 8) final: out = U2 @ out_w^T  (M=8192, N=1024, K=512) -> d_out fp32
    {
        int ntiles = DD / 64;                              // 16
        gemm_bf16_kernel<<<(BS / 128) * ntiles, 256, 0, stream>>>(
            U2, OUTW16, outp, BS, DD, LRc, ntiles);
    }
}
